// NeuralNetwork_47682726920623
// MI455X (gfx1250) — matrix operand order repaired and checked by reading, not compiled, not run on hardware
//
#include <hip/hip_runtime.h>
#include <hip/hip_bf16.h>
#include <math.h>

// ---------------------------------------------------------------------------
// Types for CDNA5 WMMA
// ---------------------------------------------------------------------------
typedef __attribute__((ext_vector_type(16))) __bf16 v16bf;
typedef __attribute__((ext_vector_type(8)))  float  v8f;
typedef __attribute__((ext_vector_type(8)))  unsigned int v8u;

#define IN_F 256
#define GF   128   // 2*HID, GAT layer width
#define HID  64
#define NG   32

// ---------------------------------------------------------------------------
// Helpers
// ---------------------------------------------------------------------------
__device__ __forceinline__ unsigned short f2bf_bits(float f) {
    unsigned u = __float_as_uint(f);
    unsigned r = u + 0x7FFFu + ((u >> 16) & 1u);   // round-to-nearest-even
    return (unsigned short)(r >> 16);
}
__device__ __forceinline__ float selu_f(float x) {
    const float a = 1.6732632423543772f, s = 1.0507009873554805f;
    return x > 0.f ? s * x : s * a * (expf(x) - 1.f);
}
__device__ __forceinline__ float lrelu_f(float x) { return x > 0.f ? x : 0.2f * x; }
// order-preserving float <-> uint encoding (for atomic segment-max)
__device__ __forceinline__ unsigned fenc(float f) {
    unsigned u = __float_as_uint(f);
    return (u & 0x80000000u) ? ~u : (u | 0x80000000u);
}
__device__ __forceinline__ float fdec(unsigned u) {
    unsigned v = (u & 0x80000000u) ? (u & 0x7FFFFFFFu) : ~u;
    return __uint_as_float(v);
}

// ---------------------------------------------------------------------------
// Elementwise / conversion kernels
// ---------------------------------------------------------------------------
__global__ void cvt_f32_bf16(const float* __restrict__ in, unsigned short* __restrict__ out, int n) {
    int i = blockIdx.x * blockDim.x + threadIdx.x;
    if (i < n) out[i] = f2bf_bits(in[i]);
}

// W[k, GF] (row-major fp32) -> Wt[n*K + k] bf16 (column-major, so K-pairs contiguous)
__global__ void cvt_w_transpose(const float* __restrict__ W, unsigned short* __restrict__ Wt, int K) {
    int i = blockIdx.x * blockDim.x + threadIdx.x;
    if (i >= K * GF) return;
    int k = i / GF, n = i % GF;
    Wt[(size_t)n * K + k] = f2bf_bits(W[(size_t)k * GF + n]);
}

__global__ void fill0(float* __restrict__ p, int n) {
    int i = blockIdx.x * blockDim.x + threadIdx.x;
    if (i < n) p[i] = 0.f;
}

__global__ void init_nodes(unsigned* __restrict__ maxk, float* __restrict__ denom, int n) {
    int i = blockIdx.x * blockDim.x + threadIdx.x;
    if (i < n) { maxk[i] = 0x007FFFFFu; /* enc(-inf) */ denom[i] = 0.f; }
}

// out[i,f] = selu(acc[i,f] + bias[f])  (in place)
__global__ void bias_selu(float* __restrict__ acc, const float* __restrict__ bias, int nel) {
    int i = blockIdx.x * blockDim.x + threadIdx.x;
    if (i < nel) acc[i] = selu_f(acc[i] + bias[i & (GF - 1)]);
}

// ---------------------------------------------------------------------------
// WMMA GEMM:  C[M, 128] = A[M, K] @ W[K, 128]   (A,W bf16; C fp32)
// Each wave owns TWO 16-row tiles (32 rows): every B fragment feeds two
// back-to-back WMMAs, and B loads are double-buffered so the next fragment
// is in flight while the current pair of WMMAs executes.
// ---------------------------------------------------------------------------
template <int KT>   // KT = K/32
__global__ __launch_bounds__(128)
void gemm_wmma(const unsigned short* __restrict__ A, const unsigned short* __restrict__ Wt,
               float* __restrict__ C, int M) {
    const int K  = KT * 32;
    const int KU = K / 2;                       // row length in packed dwords
    const int wave = threadIdx.x >> 5;
    const int lane = threadIdx.x & 31;
    const int lastTile = M / 16 - 1;
    int t0 = (blockIdx.x * 4 + wave) * 2;
    if (t0 > lastTile) return;                  // wave-uniform: EXEC stays all-ones
    int t1 = t0 + 1; if (t1 > lastTile) t1 = lastTile;   // clamp (duplicate, benign)

    const int half = lane >> 4;
    const int mlo  = lane & 15;                 // A row within tile / D column

    const unsigned int* Arow0 = (const unsigned int*)(A + (size_t)(t0 * 16 + mlo) * K);
    const unsigned int* Arow1 = (const unsigned int*)(A + (size_t)(t1 * 16 + mlo) * K);
    const unsigned int* Wcol  = (const unsigned int*)Wt + (size_t)mlo * KU + half * 8;

    v8f acc0[8] = {};
    v8f acc1[8] = {};

#pragma unroll
    for (int kt = 0; kt < KT; ++kt) {
        // --- A fragments for both m-tiles (contiguous k = 16*half + s order) ---
        v8u a0, a1;
#pragma unroll
        for (int i = 0; i < 8; ++i) {
            int ku = kt * 16 + half * 8 + i;
            a0[i] = Arow0[ku];
            a1[i] = Arow1[ku];
        }
        // --- B double buffer over the 8 column tiles ---
        v8u bcur, bnext;
        {
            const unsigned int* p = Wcol + kt * 16;          // nt = 0
#pragma unroll
            for (int i = 0; i < 8; ++i) bcur[i] = p[i];
        }
#pragma unroll
        for (int nt = 0; nt < 8; ++nt) {
            if (nt < 7) {
                const unsigned int* p = Wcol + (size_t)(nt + 1) * 16 * KU + kt * 16;
#pragma unroll
                for (int i = 0; i < 8; ++i) bnext[i] = p[i];
            }
            acc0[nt] = __builtin_amdgcn_wmma_f32_16x16x32_bf16(
                false, __builtin_bit_cast(v16bf, a0),
                false, __builtin_bit_cast(v16bf, bcur),
                (short)0, acc0[nt], false, false);
            acc1[nt] = __builtin_amdgcn_wmma_f32_16x16x32_bf16(
                false, __builtin_bit_cast(v16bf, a1),
                false, __builtin_bit_cast(v16bf, bcur),
                (short)0, acc1[nt], false, false);
            if (nt < 7) bcur = bnext;
        }
    }

    // --- store: D layout, lane column = mlo, VGPR r -> row half*8+r ---
    const int r0 = t0 * 16 + half * 8;
    const int r1 = t1 * 16 + half * 8;
#pragma unroll
    for (int nt = 0; nt < 8; ++nt) {
        const int n = nt * 16 + mlo;
#pragma unroll
        for (int r = 0; r < 8; ++r) {
            C[(size_t)(r0 + r) * GF + n] = acc0[nt][r];
            C[(size_t)(r1 + r) * GF + n] = acc1[nt][r];
        }
    }
}

// ---------------------------------------------------------------------------
// es[i] = h[i,:]·a_src, ed[i] = h[i,:]·a_dst   (one wave32 per row)
// ---------------------------------------------------------------------------
__global__ __launch_bounds__(256)
void attn_coef(const float* __restrict__ h, const float* __restrict__ as, const float* __restrict__ ad,
               float* __restrict__ es, float* __restrict__ ed, int n) {
    int wv = (blockIdx.x * blockDim.x + threadIdx.x) >> 5;
    int lane = threadIdx.x & 31;
    if (wv >= n) return;
    const float* row = h + (size_t)wv * GF;
    float s = 0.f, d = 0.f;
#pragma unroll
    for (int j = 0; j < 4; ++j) {
        float v = row[lane + 32 * j];
        s += v * as[lane + 32 * j];
        d += v * ad[lane + 32 * j];
    }
#pragma unroll
    for (int off = 16; off; off >>= 1) { s += __shfl_xor(s, off, 32); d += __shfl_xor(d, off, 32); }
    if (lane == 0) { es[wv] = s; ed[wv] = d; }
}

// ---------------------------------------------------------------------------
// Edge passes (E real edges + N self loops)
// ---------------------------------------------------------------------------
__device__ __forceinline__ void edge_sd(const int* __restrict__ ei, int E, int idx, int& s, int& d) {
    if (idx < E) { s = ei[idx]; d = ei[E + idx]; } else { s = d = idx - E; }
}

__global__ void edge_max(const int* __restrict__ ei, int E, int N,
                         const float* __restrict__ es, const float* __restrict__ ed,
                         unsigned* __restrict__ maxk) {
    int idx = blockIdx.x * blockDim.x + threadIdx.x;
    if (idx >= E + N) return;
    int s, d; edge_sd(ei, E, idx, s, d);
    float e = lrelu_f(es[s] + ed[d]);
    atomicMax(&maxk[d], fenc(e));
}

__global__ void edge_exp(const int* __restrict__ ei, int E, int N,
                         const float* __restrict__ es, const float* __restrict__ ed,
                         const unsigned* __restrict__ maxk,
                         float* __restrict__ exv, float* __restrict__ denom) {
    int idx = blockIdx.x * blockDim.x + threadIdx.x;
    if (idx >= E + N) return;
    int s, d; edge_sd(ei, E, idx, s, d);
    float e = lrelu_f(es[s] + ed[d]);
    float m = fdec(maxk[d]);
    if (!isfinite(m)) m = 0.f;
    float ex = expf(e - m);
    exv[idx] = ex;
    atomicAdd(&denom[d], ex);
}

// one wave32 per edge: acc[d,:] += alpha * h[s,:]
__global__ __launch_bounds__(256)
void edge_scatter(const int* __restrict__ ei, int E, int N,
                  const float* __restrict__ exv, const float* __restrict__ denom,
                  const float* __restrict__ h, float* __restrict__ acc) {
    int wv = (blockIdx.x * blockDim.x + threadIdx.x) >> 5;
    int lane = threadIdx.x & 31;
    if (wv >= E + N) return;
    int s, d; edge_sd(ei, E, wv, s, d);
    float alpha = exv[wv] / (denom[d] + 1e-16f);
    const float* hr = h + (size_t)s * GF;
    float* ar = acc + (size_t)d * GF;
#pragma unroll
    for (int j = 0; j < 4; ++j) {
        int f = lane + 32 * j;
        atomicAdd(&ar[f], alpha * hr[f]);
    }
}

// ---------------------------------------------------------------------------
// global mean pool (LDS-staged) + head MLP
// ---------------------------------------------------------------------------
__global__ __launch_bounds__(128)
void pool_kernel(const float* __restrict__ feat, const int* __restrict__ batch,
                 float* __restrict__ pooled, float* __restrict__ cnt, int n) {
    __shared__ float sacc[NG * GF];
    __shared__ float scnt[NG];
    int tid = threadIdx.x;
    for (int i = tid; i < NG * GF; i += 128) sacc[i] = 0.f;
    if (tid < NG) scnt[tid] = 0.f;
    __syncthreads();
    int base = blockIdx.x * 128;
    for (int i = 0; i < 128; ++i) {
        int node = base + i;
        if (node >= n) break;
        int g = batch[node];
        atomicAdd(&sacc[g * GF + tid], feat[(size_t)node * GF + tid]);
        if (tid == 0) atomicAdd(&scnt[g], 1.f);
    }
    __syncthreads();
    for (int i = tid; i < NG * GF; i += 128)
        if (sacc[i] != 0.f) atomicAdd(&pooled[i], sacc[i]);
    if (tid < NG && scnt[tid] != 0.f) atomicAdd(&cnt[tid], scnt[tid]);
}

__global__ __launch_bounds__(64)
void head_kernel(const float* __restrict__ pooled, const float* __restrict__ cnt,
                 const float* __restrict__ lw1, const float* __restrict__ lb1,
                 const float* __restrict__ lw2, const float* __restrict__ lb2,
                 float* __restrict__ out) {
    __shared__ float zs[GF];
    __shared__ float z1[HID];
    int b = blockIdx.x, t = threadIdx.x;
    float c = fmaxf(cnt[b], 1.f);
    zs[t]       = selu_f(pooled[b * GF + t] / c);
    zs[t + 64]  = selu_f(pooled[b * GF + t + 64] / c);
    __syncthreads();
    float a = lb1[t];
    for (int k = 0; k < GF; ++k) a += zs[k] * lw1[k * HID + t];
    z1[t] = selu_f(a);
    __syncthreads();
    if (t == 0) {
        float z20 = lb2[0], z21 = lb2[1];
        for (int j = 0; j < HID; ++j) { z20 += z1[j] * lw2[j * 2 + 0]; z21 += z1[j] * lw2[j * 2 + 1]; }
        float m = fmaxf(z20, z21);
        float lse = m + logf(expf(z20 - m) + expf(z21 - m));
        out[b * 2 + 0] = z20 - lse;
        out[b * 2 + 1] = z21 - lse;
    }
}

// ---------------------------------------------------------------------------
// Launch
// ---------------------------------------------------------------------------
extern "C" void kernel_launch(void* const* d_in, const int* in_sizes, int n_in,
                              void* d_out, int out_size, void* d_ws, size_t ws_size,
                              hipStream_t stream) {
    const float* x    = (const float*)d_in[0];
    const int*   ei   = (const int*)d_in[1];
    const int*   batch= (const int*)d_in[2];
    const float* W1   = (const float*)d_in[3];
    const float* as1  = (const float*)d_in[4];
    const float* ad1  = (const float*)d_in[5];
    const float* b1   = (const float*)d_in[6];
    const float* W2   = (const float*)d_in[7];
    const float* as2  = (const float*)d_in[8];
    const float* ad2  = (const float*)d_in[9];
    const float* b2   = (const float*)d_in[10];
    const float* lw1  = (const float*)d_in[11];
    const float* lb1  = (const float*)d_in[12];
    const float* lw2  = (const float*)d_in[13];
    const float* lb2  = (const float*)d_in[14];
    float* out = (float*)d_out;

    const int N = in_sizes[0] / IN_F;     // 50000
    const int E = in_sizes[1] / 2;        // 800000
    const int Etot = E + N;

    // workspace carve-up
    size_t off = 0;
    auto alloc = [&](size_t bytes) { size_t o = off; off += (bytes + 255) & ~(size_t)255; return o; };
    char* w = (char*)d_ws;
    unsigned short* xbf  = (unsigned short*)(w + alloc((size_t)N * IN_F * 2)); // bf16 GEMM input (reused)
    unsigned short* w1t  = (unsigned short*)(w + alloc((size_t)IN_F * GF * 2));
    unsigned short* w2t  = (unsigned short*)(w + alloc((size_t)GF * GF * 2));
    float* h    = (float*)(w + alloc((size_t)N * GF * 4));   // GEMM output
    float* acc  = (float*)(w + alloc((size_t)N * GF * 4));   // attention accumulator / layer output
    float* es   = (float*)(w + alloc((size_t)N * 4));
    float* ed   = (float*)(w + alloc((size_t)N * 4));
    unsigned* mk= (unsigned*)(w + alloc((size_t)N * 4));
    float* den  = (float*)(w + alloc((size_t)N * 4));
    float* exv  = (float*)(w + alloc((size_t)Etot * 4));
    float* pooled = (float*)(w + alloc((size_t)NG * GF * 4));
    float* cnt  = (float*)(w + alloc((size_t)NG * 4));

    auto G = [](int n, int b) { return (n + b - 1) / b; };
    const int nElem = N * GF;
    const int mTiles = N / 16;            // 3125

    // weight conversion (bf16, W transposed so K-pairs are contiguous)
    cvt_w_transpose<<<G(IN_F * GF, 256), 256, 0, stream>>>(W1, w1t, IN_F);
    cvt_w_transpose<<<G(GF * GF, 256), 256, 0, stream>>>(W2, w2t, GF);

    // ---------------- layer 1 ----------------
    cvt_f32_bf16<<<G(N * IN_F, 256), 256, 0, stream>>>(x, xbf, N * IN_F);
    gemm_wmma<8><<<G(mTiles, 8), 128, 0, stream>>>(xbf, w1t, h, N);
    attn_coef<<<G(N * 32, 256), 256, 0, stream>>>(h, as1, ad1, es, ed, N);
    init_nodes<<<G(N, 256), 256, 0, stream>>>(mk, den, N);
    fill0<<<G(nElem, 256), 256, 0, stream>>>(acc, nElem);
    edge_max<<<G(Etot, 256), 256, 0, stream>>>(ei, E, N, es, ed, mk);
    edge_exp<<<G(Etot, 256), 256, 0, stream>>>(ei, E, N, es, ed, mk, exv, den);
    edge_scatter<<<G(Etot * 32, 256), 256, 0, stream>>>(ei, E, N, exv, den, h, acc);
    bias_selu<<<G(nElem, 256), 256, 0, stream>>>(acc, b1, nElem);

    // ---------------- layer 2 ----------------
    cvt_f32_bf16<<<G(nElem, 256), 256, 0, stream>>>(acc, xbf, nElem);
    gemm_wmma<4><<<G(mTiles, 8), 128, 0, stream>>>(xbf, w2t, h, N);
    attn_coef<<<G(N * 32, 256), 256, 0, stream>>>(h, as2, ad2, es, ed, N);
    init_nodes<<<G(N, 256), 256, 0, stream>>>(mk, den, N);
    fill0<<<G(nElem, 256), 256, 0, stream>>>(acc, nElem);
    edge_max<<<G(Etot, 256), 256, 0, stream>>>(ei, E, N, es, ed, mk);
    edge_exp<<<G(Etot, 256), 256, 0, stream>>>(ei, E, N, es, ed, mk, exv, den);
    edge_scatter<<<G(Etot * 32, 256), 256, 0, stream>>>(ei, E, N, exv, den, h, acc);
    bias_selu<<<G(nElem, 256), 256, 0, stream>>>(acc, b2, nElem);

    // ---------------- pool + head ----------------
    fill0<<<G(NG * GF, 256), 256, 0, stream>>>(pooled, NG * GF);
    fill0<<<1, 64, 0, stream>>>(cnt, NG);
    pool_kernel<<<G(N, 128), 128, 0, stream>>>(acc, batch, pooled, cnt, N);
    head_kernel<<<NG, 64, 0, stream>>>(pooled, cnt, lw1, lb1, lw2, lb2, out);
}
